// MultiHeadAttention_35854386987415
// MI455X (gfx1250) — matrix operand order repaired and checked by reading, not compiled, not run on hardware
//
#include <hip/hip_runtime.h>
#include <hip/hip_bf16.h>

// Problem constants (fixed by the reference)
#define DMODEL 1024
#define NHEADS 16
#define DKV    64
#define BATCH  2
#define SEQ    2048
#define LN_EPS 1e-6f

typedef __attribute__((ext_vector_type(16))) _Float16 v16h;
typedef __attribute__((ext_vector_type(8)))  _Float16 h8;
typedef __attribute__((ext_vector_type(4)))  _Float16 h4;
typedef __attribute__((ext_vector_type(2)))  _Float16 h2;
typedef __attribute__((ext_vector_type(8)))  float    v8f;

// LDS pitch (halves). 40 halves = 80 B/row: 16-B aligned rows, and the 16
// lanes' b128 reads hit 16 distinct 16-B chunks mod 256 B -> conflict-free.
#define PCH 40

// ---------------------------------------------------------------------------
// WMMA helper: CDNA5 v_wmma_f32_16x16x32_f16 (wave32)
// ---------------------------------------------------------------------------
__device__ __forceinline__ v8f wmma_f16(v16h a, v16h b, v8f c) {
  return __builtin_amdgcn_wmma_f32_16x16x32_f16(false, a, false, b,
                                                (short)0, c, false, false);
}

// A fragment 16x32 f16 from k-contiguous LDS tile As[m][k] (pitch PCH).
// lane L: row m = L&15, kb = 8*(L>>4); halves = A[m][kb..kb+7] ++ A[m][kb+16..kb+23]
__device__ __forceinline__ v16h load_frag_a(const _Float16* As, int m0, int lane) {
  const _Float16* p = As + (m0 + (lane & 15)) * PCH + ((lane >> 4) << 3);
  const h8 lo = *(const h8*)(p);
  const h8 hi = *(const h8*)(p + 16);
  return __builtin_shufflevector(lo, hi, 0,1,2,3,4,5,6,7,8,9,10,11,12,13,14,15);
}

// B fragment 32x16 f16 from n-major LDS tile Bs[n][k] (pitch PCH).
// lane L: col n = L&15, kb = 8*(L>>4); halves = B[kb..kb+7][n] ++ B[kb+16..kb+23][n]
__device__ __forceinline__ v16h load_frag_b(const _Float16* Bs, int n0, int lane) {
  const _Float16* p = Bs + (n0 + (lane & 15)) * PCH + ((lane >> 4) << 3);
  const h8 lo = *(const h8*)(p);
  const h8 hi = *(const h8*)(p + 16);
  return __builtin_shufflevector(lo, hi, 0,1,2,3,4,5,6,7,8,9,10,11,12,13,14,15);
}

// ---------------------------------------------------------------------------
// Generic dense projection GEMM: C[M,N] = A[M,K] * W[K,N] + bias
// Block = 256 thr (8 waves); block tile 128x64; wave patch 32x32 (4 WMMA).
// OUT_MODE 0: row-major [M,N]; OUT_MODE 1: scatter to [B,H,S,64].
// ---------------------------------------------------------------------------
template <int OUT_MODE>
__global__ __launch_bounds__(256)
void gemm_proj_kernel(const float* __restrict__ A, const float* __restrict__ W,
                      const float* __restrict__ bias, float* __restrict__ out,
                      int M, int N, int K) {
  __shared__ _Float16 As[128 * PCH];   // 128 rows x 32 k
  __shared__ _Float16 Bs[64 * PCH];    // 64 n-rows x 32 k (n-major!)
  const int tid  = threadIdx.x;
  const int lane = tid & 31;
  const int wv   = tid >> 5;
  const int wr   = wv >> 1;            // 0..3 : 32-row band
  const int wc   = wv & 1;             // 0..1 : 32-col band
  const int row0 = blockIdx.y * 128;
  const int col0 = blockIdx.x * 64;

  v8f acc[2][2];
#pragma unroll
  for (int i = 0; i < 2; ++i)
#pragma unroll
    for (int j = 0; j < 2; ++j)
      acc[i][j] = (v8f){0.f,0.f,0.f,0.f,0.f,0.f,0.f,0.f};

  // B staging microtile coords: 2(k) x 4(n) per thread
  const int kq = tid >> 4;             // 0..15 -> k-pair
  const int nq = tid & 15;             // 0..15 -> n-quad

  for (int k0 = 0; k0 < K; k0 += 32) {
    __syncthreads();
    // Stage A 128x32 fp32->f16 (1024 float4; 4 per thread, coalesced)
#pragma unroll
    for (int j = 0; j < 4; ++j) {
      const int idx4 = tid + 256 * j;
      const int r = idx4 >> 3, c = (idx4 & 7) * 4;
      const float4 v = *(const float4*)(A + (size_t)(row0 + r) * K + k0 + c);
      *(h4*)&As[r * PCH + c] =
          (h4){(_Float16)v.x, (_Float16)v.y, (_Float16)v.z, (_Float16)v.w};
    }
    // Stage B 32x64 -> n-major with 2x4 register transpose
    {
      const float* wp = W + (size_t)(k0 + 2 * kq) * N + col0 + 4 * nq;
      const float4 v0 = *(const float4*)(wp);
      const float4 v1 = *(const float4*)(wp + N);
      _Float16* bp = &Bs[(4 * nq) * PCH + 2 * kq];
      *(h2*)(bp + 0 * PCH) = (h2){(_Float16)v0.x, (_Float16)v1.x};
      *(h2*)(bp + 1 * PCH) = (h2){(_Float16)v0.y, (_Float16)v1.y};
      *(h2*)(bp + 2 * PCH) = (h2){(_Float16)v0.z, (_Float16)v1.z};
      *(h2*)(bp + 3 * PCH) = (h2){(_Float16)v0.w, (_Float16)v1.w};
    }
    // Pull next K-tile toward the WGP while WMMAs run
    __builtin_prefetch(A + (size_t)(row0 + (tid >> 1)) * K + k0 + 32, 0, 1);
    __builtin_prefetch(W + (size_t)(k0 + 32 + (tid >> 3)) * N + col0, 0, 1);
    __syncthreads();

    v16h af[2], bf[2];
#pragma unroll
    for (int i = 0; i < 2; ++i) af[i] = load_frag_a(As, wr * 32 + i * 16, lane);
#pragma unroll
    for (int j = 0; j < 2; ++j) bf[j] = load_frag_b(Bs, wc * 32 + j * 16, lane);
#pragma unroll
    for (int i = 0; i < 2; ++i)
#pragma unroll
      for (int j = 0; j < 2; ++j)
        acc[i][j] = wmma_f16(af[i], bf[j], acc[i][j]);
  }

  // Epilogue (branch-free; bias always present)
  const int n  = lane & 15;
  const int mb = (lane >> 4) << 3;
#pragma unroll
  for (int ti = 0; ti < 2; ++ti) {
#pragma unroll
    for (int tj = 0; tj < 2; ++tj) {
      const int gc = col0 + wc * 32 + tj * 16 + n;
      const float bv = bias[gc];
      const int gr0 = row0 + wr * 32 + ti * 16 + mb;
#pragma unroll
      for (int i = 0; i < 8; ++i) {
        const int gr = gr0 + i;
        const float val = acc[ti][tj][i] + bv;
        if (OUT_MODE == 0) {
          out[(size_t)gr * N + gc] = val;
        } else {
          const int b = gr >> 11, s = gr & (SEQ - 1);
          const int h = gc >> 6,  d = gc & 63;
          out[(((size_t)b * NHEADS + h) * SEQ + s) * DKV + d] = val;
        }
      }
    }
  }
}

// ---------------------------------------------------------------------------
// Attention scores: attn[bh,q,k] = (Qp[bh] . Kp[bh]^T)/8, masked (pre-softmax)
// Qp,Kp: [B*H, S, 64]. Block tile 128(q) x 64(k); B staging is a plain row
// copy of Kp (n-major layout == Kp row-major). K-dim = 64 (2 wmma steps).
// ---------------------------------------------------------------------------
__global__ __launch_bounds__(256)
void attn_scores_kernel(const float* __restrict__ qp, const float* __restrict__ kp,
                        const unsigned char* __restrict__ mask,
                        float* __restrict__ attn) {
  __shared__ _Float16 As[128 * PCH];
  __shared__ _Float16 Bs[64 * PCH];
  const int tid  = threadIdx.x;
  const int lane = tid & 31;
  const int wv   = tid >> 5;
  const int wr   = wv >> 1;
  const int wc   = wv & 1;
  const int col0 = blockIdx.x * 64;     // key positions
  const int row0 = blockIdx.y * 128;    // query positions
  const int bh   = blockIdx.z;
  const float* Qb = qp + (size_t)bh * SEQ * DKV;
  const float* Kb = kp + (size_t)bh * SEQ * DKV;

  v8f acc[2][2];
#pragma unroll
  for (int i = 0; i < 2; ++i)
#pragma unroll
    for (int j = 0; j < 2; ++j)
      acc[i][j] = (v8f){0.f,0.f,0.f,0.f,0.f,0.f,0.f,0.f};

  for (int k0 = 0; k0 < DKV; k0 += 32) {
    __syncthreads();
    // Stage Q tile 128x32
#pragma unroll
    for (int j = 0; j < 4; ++j) {
      const int idx4 = tid + 256 * j;
      const int r = idx4 >> 3, c = (idx4 & 7) * 4;
      const float4 v = *(const float4*)(Qb + (size_t)(row0 + r) * DKV + k0 + c);
      *(h4*)&As[r * PCH + c] =
          (h4){(_Float16)v.x, (_Float16)v.y, (_Float16)v.z, (_Float16)v.w};
    }
    // Stage K tile: Bs[n][k] = Kp[col0+n][k0+k]  (direct row copy, no transpose)
#pragma unroll
    for (int j = 0; j < 2; ++j) {
      const int idx4 = tid + 256 * j;
      const int rn = idx4 >> 3, c = (idx4 & 7) * 4;
      const float4 v = *(const float4*)(Kb + (size_t)(col0 + rn) * DKV + k0 + c);
      *(h4*)&Bs[rn * PCH + c] =
          (h4){(_Float16)v.x, (_Float16)v.y, (_Float16)v.z, (_Float16)v.w};
    }
    __syncthreads();

    v16h af[2], bf[2];
#pragma unroll
    for (int i = 0; i < 2; ++i) af[i] = load_frag_a(As, wr * 32 + i * 16, lane);
#pragma unroll
    for (int j = 0; j < 2; ++j) bf[j] = load_frag_b(Bs, wc * 32 + j * 16, lane);
#pragma unroll
    for (int i = 0; i < 2; ++i)
#pragma unroll
      for (int j = 0; j < 2; ++j)
        acc[i][j] = wmma_f16(af[i], bf[j], acc[i][j]);
  }

  const int n  = lane & 15;
  const int mb = (lane >> 4) << 3;
  const int bidx = bh >> 4;            // batch (H=16)
  const size_t abase = (size_t)bh * SEQ * SEQ;
  const float scale = 0.125f;          // 1/sqrt(64)
#pragma unroll
  for (int ti = 0; ti < 2; ++ti) {
#pragma unroll
    for (int tj = 0; tj < 2; ++tj) {
      const int gk = col0 + wc * 32 + tj * 16 + n;
#pragma unroll
      for (int i = 0; i < 8; ++i) {
        const int gq = row0 + wr * 32 + ti * 16 + mb + i;
        const unsigned char m = mask[((size_t)bidx * SEQ + gq) * SEQ + gk];
        attn[abase + (size_t)gq * SEQ + gk] =
            m ? -1e9f : acc[ti][tj][i] * scale;
      }
    }
  }
}

// ---------------------------------------------------------------------------
// Row softmax in place over attn rows of length SEQ=2048. One wave per row.
// ---------------------------------------------------------------------------
__global__ __launch_bounds__(256)
void softmax_kernel(float* __restrict__ attn) {
  const int lane = threadIdx.x & 31;
  const int wv   = threadIdx.x >> 5;
  const size_t row = (size_t)blockIdx.x * 8 + wv;
  float* p = attn + row * SEQ;

  float4 v[16];
  float mx = -3.402823e38f;
#pragma unroll
  for (int i = 0; i < 16; ++i) {
    v[i] = ((const float4*)p)[lane + i * 32];
    mx = fmaxf(mx, fmaxf(fmaxf(v[i].x, v[i].y), fmaxf(v[i].z, v[i].w)));
  }
#pragma unroll
  for (int off = 16; off > 0; off >>= 1)
    mx = fmaxf(mx, __shfl_xor(mx, off, 32));

  float sum = 0.f;
#pragma unroll
  for (int i = 0; i < 16; ++i) {
    v[i].x = __expf(v[i].x - mx); v[i].y = __expf(v[i].y - mx);
    v[i].z = __expf(v[i].z - mx); v[i].w = __expf(v[i].w - mx);
    sum += v[i].x + v[i].y + v[i].z + v[i].w;
  }
#pragma unroll
  for (int off = 16; off > 0; off >>= 1)
    sum += __shfl_xor(sum, off, 32);

  const float inv = 1.f / sum;
#pragma unroll
  for (int i = 0; i < 16; ++i) {
    v[i].x *= inv; v[i].y *= inv; v[i].z *= inv; v[i].w *= inv;
    ((float4*)p)[lane + i * 32] = v[i];
  }
}

// ---------------------------------------------------------------------------
// Context GEMM: ctx[b,s,h,d] = sum_k attn[bh,s,k] * Vp[bh,k,d]
// Per bh: [2048,64] = [2048,2048] x [2048,64]. Block tile 128x64 (full N).
// ---------------------------------------------------------------------------
__global__ __launch_bounds__(256)
void attn_context_kernel(const float* __restrict__ attn,
                         const float* __restrict__ vp,
                         float* __restrict__ ctx) {
  __shared__ _Float16 As[128 * PCH];
  __shared__ _Float16 Bs[64 * PCH];
  const int tid  = threadIdx.x;
  const int lane = tid & 31;
  const int wv   = tid >> 5;
  const int wr   = wv >> 1;
  const int wc   = wv & 1;
  const int row0 = blockIdx.x * 128;
  const int bh   = blockIdx.y;
  const float* Ab = attn + (size_t)bh * SEQ * SEQ;
  const float* Vb = vp   + (size_t)bh * SEQ * DKV;

  v8f acc[2][2];
#pragma unroll
  for (int i = 0; i < 2; ++i)
#pragma unroll
    for (int j = 0; j < 2; ++j)
      acc[i][j] = (v8f){0.f,0.f,0.f,0.f,0.f,0.f,0.f,0.f};

  const int kq = tid >> 4;
  const int nq = tid & 15;

  for (int k0 = 0; k0 < SEQ; k0 += 32) {
    __syncthreads();
    // Stage attn tile 128x32
#pragma unroll
    for (int j = 0; j < 4; ++j) {
      const int idx4 = tid + 256 * j;
      const int r = idx4 >> 3, c = (idx4 & 7) * 4;
      const float4 v = *(const float4*)(Ab + (size_t)(row0 + r) * SEQ + k0 + c);
      *(h4*)&As[r * PCH + c] =
          (h4){(_Float16)v.x, (_Float16)v.y, (_Float16)v.z, (_Float16)v.w};
    }
    // Stage V tile 32x64 -> n-major (2x4 register transpose)
    {
      const float* wp = Vb + (size_t)(k0 + 2 * kq) * DKV + 4 * nq;
      const float4 v0 = *(const float4*)(wp);
      const float4 v1 = *(const float4*)(wp + DKV);
      _Float16* bp = &Bs[(4 * nq) * PCH + 2 * kq];
      *(h2*)(bp + 0 * PCH) = (h2){(_Float16)v0.x, (_Float16)v1.x};
      *(h2*)(bp + 1 * PCH) = (h2){(_Float16)v0.y, (_Float16)v1.y};
      *(h2*)(bp + 2 * PCH) = (h2){(_Float16)v0.z, (_Float16)v1.z};
      *(h2*)(bp + 3 * PCH) = (h2){(_Float16)v0.w, (_Float16)v1.w};
    }
    __builtin_prefetch(Ab + (size_t)(row0 + (tid >> 1)) * SEQ + k0 + 32, 0, 1);
    __builtin_prefetch(Vb + (size_t)(k0 + 32 + (tid >> 3)) * DKV, 0, 1);
    __syncthreads();

    v16h af[2], bf[2];
#pragma unroll
    for (int i = 0; i < 2; ++i) af[i] = load_frag_a(As, wr * 32 + i * 16, lane);
#pragma unroll
    for (int j = 0; j < 2; ++j) bf[j] = load_frag_b(Bs, wc * 32 + j * 16, lane);
#pragma unroll
    for (int i = 0; i < 2; ++i)
#pragma unroll
      for (int j = 0; j < 2; ++j)
        acc[i][j] = wmma_f16(af[i], bf[j], acc[i][j]);
  }

  const int n  = lane & 15;
  const int mb = (lane >> 4) << 3;
  const int b  = bh >> 4, h = bh & 15;
#pragma unroll
  for (int ti = 0; ti < 2; ++ti) {
#pragma unroll
    for (int tj = 0; tj < 2; ++tj) {
      const int d = wc * 32 + tj * 16 + n;   // 0..63
#pragma unroll
      for (int i = 0; i < 8; ++i) {
        const int s = row0 + wr * 32 + ti * 16 + mb + i;
        ctx[(((size_t)b * SEQ + s) * NHEADS + h) * DKV + d] = acc[ti][tj][i];
      }
    }
  }
}

// ---------------------------------------------------------------------------
// Residual + LayerNorm: y = LN(proj + Q) * gamma + beta. One block per row.
// ---------------------------------------------------------------------------
__global__ __launch_bounds__(256)
void residual_ln_kernel(const float* __restrict__ proj,
                        const float* __restrict__ Qin,
                        const float* __restrict__ gamma,
                        const float* __restrict__ beta,
                        float* __restrict__ y) {
  __shared__ float red_s[8];
  __shared__ float red_q[8];
  const int row = blockIdx.x;
  const int i0  = threadIdx.x * 4;
  const float* pr = proj + (size_t)row * DMODEL;
  const float* qr = Qin  + (size_t)row * DMODEL;

  const float4 a = *(const float4*)(pr + i0);
  const float4 b = *(const float4*)(qr + i0);
  const float x0 = a.x + b.x, x1 = a.y + b.y, x2 = a.z + b.z, x3 = a.w + b.w;
  float s  = x0 + x1 + x2 + x3;
  float s2 = x0 * x0 + x1 * x1 + x2 * x2 + x3 * x3;
#pragma unroll
  for (int off = 16; off > 0; off >>= 1) {
    s  += __shfl_xor(s,  off, 32);
    s2 += __shfl_xor(s2, off, 32);
  }
  const int lane = threadIdx.x & 31, wv = threadIdx.x >> 5;
  if (lane == 0) { red_s[wv] = s; red_q[wv] = s2; }
  __syncthreads();
  float ts = 0.f, ts2 = 0.f;
#pragma unroll
  for (int i = 0; i < 8; ++i) { ts += red_s[i]; ts2 += red_q[i]; }
  const float mu  = ts * (1.f / DMODEL);
  const float var = ts2 * (1.f / DMODEL) - mu * mu;
  const float inv = rsqrtf(var + LN_EPS);

  const float4 g  = *(const float4*)(gamma + i0);
  const float4 be = *(const float4*)(beta + i0);
  float4 o;
  o.x = (x0 - mu) * inv * g.x + be.x;
  o.y = (x1 - mu) * inv * g.y + be.y;
  o.z = (x2 - mu) * inv * g.z + be.z;
  o.w = (x3 - mu) * inv * g.w + be.w;
  *(float4*)(y + (size_t)row * DMODEL + i0) = o;
}

// ---------------------------------------------------------------------------
extern "C" void kernel_launch(void* const* d_in, const int* in_sizes, int n_in,
                              void* d_out, int out_size, void* d_ws, size_t ws_size,
                              hipStream_t stream) {
  (void)in_sizes; (void)n_in; (void)out_size; (void)ws_size;
  const float* Q    = (const float*)d_in[0];
  const float* Kin  = (const float*)d_in[1];
  const float* Vin  = (const float*)d_in[2];
  const unsigned char* mask = (const unsigned char*)d_in[3];
  const float* W_q = (const float*)d_in[4];  const float* b_q = (const float*)d_in[5];
  const float* W_k = (const float*)d_in[6];  const float* b_k = (const float*)d_in[7];
  const float* W_v = (const float*)d_in[8];  const float* b_v = (const float*)d_in[9];
  const float* W_o = (const float*)d_in[10]; const float* b_o = (const float*)d_in[11];
  const float* gamma = (const float*)d_in[12];
  const float* beta  = (const float*)d_in[13];

  float* y    = (float*)d_out;
  float* attn = y + (size_t)BATCH * SEQ * DMODEL;  // outputs: (y, attn) flat

  const size_t PROJ = (size_t)BATCH * SEQ * NHEADS * DKV;  // 4,194,304 floats
  float* ws   = (float*)d_ws;
  float* qp   = ws;             // [B,H,S,64]
  float* kp   = ws + PROJ;      // [B,H,S,64]
  float* vp   = ws + 2 * PROJ;  // [B,H,S,64]
  float* ctx  = ws + 3 * PROJ;  // [B,S,H,64]
  float* proj = ws + 4 * PROJ;  // [B,S,DMODEL]

  const int M = BATCH * SEQ;    // 4096
  dim3 blk(256);
  dim3 gProj(DMODEL / 64, M / 128);                // (16, 32)

  gemm_proj_kernel<1><<<gProj, blk, 0, stream>>>(Q,   W_q, b_q, qp, M, DMODEL, DMODEL);
  gemm_proj_kernel<1><<<gProj, blk, 0, stream>>>(Kin, W_k, b_k, kp, M, DMODEL, DMODEL);
  gemm_proj_kernel<1><<<gProj, blk, 0, stream>>>(Vin, W_v, b_v, vp, M, DMODEL, DMODEL);

  dim3 gScores(SEQ / 64, SEQ / 128, BATCH * NHEADS);  // (32, 16, 32)
  attn_scores_kernel<<<gScores, blk, 0, stream>>>(qp, kp, mask, attn);

  softmax_kernel<<<(BATCH * NHEADS * SEQ) / 8, blk, 0, stream>>>(attn);

  dim3 gCtx(SEQ / 128, BATCH * NHEADS);               // (16, 32)
  attn_context_kernel<<<gCtx, blk, 0, stream>>>(attn, vp, ctx);

  gemm_proj_kernel<0><<<gProj, blk, 0, stream>>>(ctx, W_o, b_o, proj, M, DMODEL, DMODEL);

  residual_ln_kernel<<<M, blk, 0, stream>>>(proj, Q, gamma, beta, y);
}
